// MultiHeadAttention_49778670961196
// MI455X (gfx1250) — matrix operand order repaired and checked by reading, not compiled, not run on hardware
//
#include <hip/hip_runtime.h>

// ---------------------------------------------------------------------------
// Problem constants (match reference)
// ---------------------------------------------------------------------------
constexpr int B   = 4;
constexpr int T   = 2048;
constexpr int C   = 1024;
constexpr int G   = 4;
constexpr int HPG = 4;
constexpr int D   = 64;
constexpr int NCOL = G * HPG * D;   // 1024, concat width

typedef __bf16 bf16_t;
typedef __attribute__((ext_vector_type(16))) __bf16 v16bf;
typedef __attribute__((ext_vector_type(8)))  __bf16 v8bf;
typedef __attribute__((ext_vector_type(8)))  float  v8f;
typedef __attribute__((ext_vector_type(4)))  int    v4i;

#if __has_builtin(__builtin_amdgcn_global_load_async_to_lds_b128) && \
    __has_builtin(__builtin_amdgcn_s_wait_asynccnt)
#define FLASH_ASYNC 1
#else
#define FLASH_ASYNC 0
#endif

// ---------------------------------------------------------------------------
// WMMA helper: D = A(16x32 bf16) * B(32x16 bf16) + C(f32)
// ---------------------------------------------------------------------------
__device__ __forceinline__ v8f wmma_bf16(v16bf a, v16bf b, v8f c) {
  return __builtin_amdgcn_wmma_f32_16x16x32_bf16(false, a, false, b,
                                                 (short)0, c, false, false);
}

// A operand, 16x32 bf16, row-major source with leading dim `ld`.
// ISA layout: lane<16 holds row=lane, K={0..7,16..23}; lane>=16 holds
// row=lane-16, K={8..15,24..31}.  -> two contiguous 16B loads per lane.
__device__ __forceinline__ v16bf load_a_16x32(const bf16_t* __restrict__ base, int ld) {
  const int lane = threadIdx.x & 31;
  const int row  = lane & 15;
  const int half = lane >> 4;
  const bf16_t* p = base + (size_t)row * ld + half * 8;
  v8bf lo = *(const v8bf*)(p);
  v8bf hi = *(const v8bf*)(p + 16);
  v16bf r;
#pragma unroll
  for (int i = 0; i < 8; ++i) { r[i] = lo[i]; r[i + 8] = hi[i]; }
  return r;
}

// B operand, 32x16 bf16.  Memory holds B^T (row n = output column, contiguous
// along K).  ISA layout: lanes 0-15 col=lane K={0..7,16..23}, lanes 16-31
// col=lane-16 K={8..15,24..31}  -> two contiguous 16B loads per lane.
__device__ __forceinline__ v16bf load_b_32x16(const bf16_t* __restrict__ base, int ld) {
  const int lane = threadIdx.x & 31;
  const int col  = lane & 15;
  const int half = lane >> 4;
  const bf16_t* p = base + (size_t)col * ld + half * 8;
  v8bf lo = *(const v8bf*)(p);
  v8bf hi = *(const v8bf*)(p + 16);
  v16bf r;
#pragma unroll
  for (int i = 0; i < 8; ++i) { r[i] = lo[i]; r[i + 8] = hi[i]; }
  return r;
}

#if FLASH_ASYNC
// one 16-byte async global->LDS copy per lane (ASYNCcnt-tracked)
__device__ __forceinline__ void async_b128(const bf16_t* g, bf16_t* l) {
  __builtin_amdgcn_global_load_async_to_lds_b128(
      (__attribute__((address_space(1))) v4i*)(g),
      (__attribute__((address_space(3))) v4i*)(l), 0, 0);
}
#endif

// ---------------------------------------------------------------------------
// Kernel 0a: fp32 -> bf16 plain convert
// ---------------------------------------------------------------------------
__global__ void cvt_kernel(const float* __restrict__ src, bf16_t* __restrict__ dst, int n) {
  int i = blockIdx.x * blockDim.x + threadIdx.x;
  if (i < n) dst[i] = (bf16_t)src[i];
}

// Kernel 0b: fp32 -> bf16 with per-matrix transpose.
// src: [nmat, rows, cols]  ->  dst: [nmat, cols, rows]
__global__ void cvt_tr_kernel(const float* __restrict__ src, bf16_t* __restrict__ dst,
                              int nmat, int rows, int cols) {
  int i = blockIdx.x * blockDim.x + threadIdx.x;
  int total = nmat * rows * cols;
  if (i >= total) return;
  int m   = i / (rows * cols);
  int rem = i - m * rows * cols;
  int r   = rem / cols;
  int c   = rem - r * cols;
  dst[(size_t)m * rows * cols + (size_t)c * rows + r] = (bf16_t)src[i];
}

// ---------------------------------------------------------------------------
// Kernel 1: fused QKV projection (register-pipelined K loop).
// One wave -> one 16(t) x 64(d) tile for one head-unit (HPG Q heads + K + V).
// Q -> [B,G,HPG,T,D], K -> [B,G,T,D], V -> [B,G,D,T] (transposed)
// ---------------------------------------------------------------------------
__global__ void __launch_bounds__(256)
qkv_proj_kernel(const bf16_t* __restrict__ xb,
                const bf16_t* __restrict__ Wqt,   // [G*HPG, D, C]
                const bf16_t* __restrict__ Wkt,   // [G, D, C]
                const bf16_t* __restrict__ Wvt,   // [G, D, C]
                bf16_t* __restrict__ Qb, bf16_t* __restrict__ Kb, bf16_t* __restrict__ Vt) {
  const int wave = threadIdx.x >> 5;
  const int lane = threadIdx.x & 31;
  const int tg   = blockIdx.x * 8 + wave;          // 12288 waves total
  const int mtile = tg & 127;                      // T/16 tiles
  const int hu    = tg >> 7;                       // 0..95
  const int b = hu / (G * (HPG + 2));
  const int rem = hu - b * (G * (HPG + 2));
  const int g = rem / (HPG + 2);
  const int u = rem - g * (HPG + 2);
  const int Mbase = mtile * 16;

  const bf16_t* wt;
  if (u < HPG)          wt = Wqt + ((size_t)(g * HPG + u) * D) * C;
  else if (u == HPG)    wt = Wkt + ((size_t)g * D) * C;
  else                  wt = Wvt + ((size_t)g * D) * C;

  const bf16_t* xrow = xb + ((size_t)b * T + Mbase) * C;

  v8f acc0 = {}, acc1 = {}, acc2 = {}, acc3 = {};
  // software-pipelined: stage kc+32 while multiplying kc
  v16bf a  = load_a_16x32(xrow, C);
  v16bf b0 = load_b_32x16(wt + (size_t)0  * C, C);
  v16bf b1 = load_b_32x16(wt + (size_t)16 * C, C);
  v16bf b2 = load_b_32x16(wt + (size_t)32 * C, C);
  v16bf b3 = load_b_32x16(wt + (size_t)48 * C, C);
  for (int kc = 32; kc < C; kc += 32) {
    v16bf an  = load_a_16x32(xrow + kc, C);
    v16bf bn0 = load_b_32x16(wt + (size_t)0  * C + kc, C);
    v16bf bn1 = load_b_32x16(wt + (size_t)16 * C + kc, C);
    v16bf bn2 = load_b_32x16(wt + (size_t)32 * C + kc, C);
    v16bf bn3 = load_b_32x16(wt + (size_t)48 * C + kc, C);
    acc0 = wmma_bf16(a, b0, acc0);
    acc1 = wmma_bf16(a, b1, acc1);
    acc2 = wmma_bf16(a, b2, acc2);
    acc3 = wmma_bf16(a, b3, acc3);
    a = an; b0 = bn0; b1 = bn1; b2 = bn2; b3 = bn3;
  }
  acc0 = wmma_bf16(a, b0, acc0);
  acc1 = wmma_bf16(a, b1, acc1);
  acc2 = wmma_bf16(a, b2, acc2);
  acc3 = wmma_bf16(a, b3, acc3);

  const int colBase = lane & 15;
  const int rowOff  = (lane >> 4) << 3;            // 0 or 8

  if (u < HPG) {                                   // Q: [b,g,u, t, d]
    bf16_t* dst = Qb + (((size_t)(b * G + g) * HPG + u) * T + Mbase) * D;
#pragma unroll
    for (int r = 0; r < 8; ++r) {
      const size_t ra = (size_t)(rowOff + r) * D;
      dst[ra + 0  + colBase] = (bf16_t)acc0[r];
      dst[ra + 16 + colBase] = (bf16_t)acc1[r];
      dst[ra + 32 + colBase] = (bf16_t)acc2[r];
      dst[ra + 48 + colBase] = (bf16_t)acc3[r];
    }
  } else if (u == HPG) {                           // K: [b,g, t, d]
    bf16_t* dst = Kb + ((size_t)(b * G + g) * T + Mbase) * D;
#pragma unroll
    for (int r = 0; r < 8; ++r) {
      const size_t ra = (size_t)(rowOff + r) * D;
      dst[ra + 0  + colBase] = (bf16_t)acc0[r];
      dst[ra + 16 + colBase] = (bf16_t)acc1[r];
      dst[ra + 32 + colBase] = (bf16_t)acc2[r];
      dst[ra + 48 + colBase] = (bf16_t)acc3[r];
    }
  } else {                                         // V transposed: [b,g, d, t]
    bf16_t* dst = Vt + ((size_t)(b * G + g) * D) * T;
    v8f* accs[4] = { &acc0, &acc1, &acc2, &acc3 };
#pragma unroll
    for (int j = 0; j < 4; ++j) {
      v8bf pk;
#pragma unroll
      for (int r = 0; r < 8; ++r) pk[r] = (bf16_t)((*accs[j])[r]);
      *(v8bf*)(dst + (size_t)(j * 16 + colBase) * T + Mbase + rowOff) = pk;
    }
  }
}

// ---------------------------------------------------------------------------
// Kernel 2: causal flash attention, async-LDS double-buffered K/V staging.
// One wave -> one 16-row query tile for one (b,g,h). Key blocks of 32.
// Per-wave LDS: K stage 2x4KB + V stage 2x4KB + P staging 1KB = 17408 B.
// ---------------------------------------------------------------------------
constexpr int FLASH_WAVE_SMEM = 17408;
constexpr int FLASH_SMEM_BYTES = 8 * FLASH_WAVE_SMEM;   // 139264 B dynamic LDS

__global__ void __launch_bounds__(256)
flash_attn_kernel(const bf16_t* __restrict__ Qb, const bf16_t* __restrict__ Kb,
                  const bf16_t* __restrict__ Vt, bf16_t* __restrict__ attnb) {
  extern __shared__ __align__(16) char smem[];

  const int wave = threadIdx.x >> 5;
  const int lane = threadIdx.x & 31;
  const int tg   = blockIdx.x * 8 + wave;          // 8192 waves
  const int mtile = tg & 127;
  const int bgh   = tg >> 7;                       // 0..63
  const int h = bgh & (HPG - 1);
  const int g = (bgh >> 2) & (G - 1);
  const int b = bgh >> 4;
  const int Mbase = mtile * 16;

  const bf16_t* qptr = Qb + (((size_t)(b * G + g) * HPG + h) * T + Mbase) * D;
  const bf16_t* kptr = Kb + ((size_t)(b * G + g) * T) * D;
  const bf16_t* vptr = Vt + ((size_t)(b * G + g) * D) * T;

#if FLASH_ASYNC
  char* wbase = smem + (size_t)wave * FLASH_WAVE_SMEM;
  bf16_t* kst[2] = { (bf16_t*)wbase,            (bf16_t*)(wbase + 4096) };
  bf16_t* vst[2] = { (bf16_t*)(wbase + 8192),   (bf16_t*)(wbase + 12288) };
  bf16_t* myp    = (bf16_t*)(wbase + 16384);
#else
  bf16_t* myp = (bf16_t*)(smem + (size_t)wave * 1024);
#endif

  const v16bf qa0 = load_a_16x32(qptr,      D);    // K = d 0..31
  const v16bf qa1 = load_a_16x32(qptr + 32, D);    // K = d 32..63

  v8f o0 = {}, o1 = {}, o2 = {}, o3 = {};
  float m_i[8], l_i[8];
#pragma unroll
  for (int r = 0; r < 8; ++r) { m_i[r] = -3.0e38f; l_i[r] = 0.f; }

  const int colBase = lane & 15;
  const int rowOff  = (lane >> 4) << 3;
  const int nblk = (Mbase >> 5) + 1;               // causal block count

#if FLASH_ASYNC
  // issue one 32-wide K/V block into LDS stage buffers (16 async b128 per lane)
  auto issue_stage = [&](int sb) {
    const int buf = sb & 1;
    const bf16_t* kg = kptr + (size_t)(sb << 5) * D;   // contiguous 4KB
    bf16_t* kl = kst[buf];
#pragma unroll
    for (int i = 0; i < 8; ++i) {
      const int e = lane * 8 + i * 256;                // 16B chunks
      async_b128(kg + e, kl + e);
    }
    const bf16_t* vg = vptr + (sb << 5);               // 64 rows x 64B, stride T
    bf16_t* vl = vst[buf];
#pragma unroll
    for (int i = 0; i < 8; ++i) {
      const int chunk = lane + i * 32;                 // 0..255
      const int row = chunk >> 2;
      const int cw  = (chunk & 3) * 8;
      async_b128(vg + (size_t)row * T + cw, vl + row * 32 + cw);
    }
  };
  issue_stage(0);
#endif

  for (int sb = 0; sb < nblk; ++sb) {
    const int s0 = sb << 5;

#if FLASH_ASYNC
    if (sb + 1 < nblk) {                // overlap next block's DMA with compute
      issue_stage(sb + 1);
      __builtin_amdgcn_s_wait_asynccnt(16);   // retire the 16 ops of block sb
    } else {
      __builtin_amdgcn_s_wait_asynccnt(0);
    }
    const bf16_t* kt = kst[sb & 1];     // [32 s][64 d] row-major in LDS
    const bf16_t* vt = vst[sb & 1];     // [64 d][32 s] row-major in LDS
    v16bf kb0 = load_b_32x16(kt,                64);
    v16bf kb1 = load_b_32x16(kt + 32,           64);
    v16bf kb2 = load_b_32x16(kt + 16 * 64,      64);
    v16bf kb3 = load_b_32x16(kt + 16 * 64 + 32, 64);
#else
    v16bf kb0 = load_b_32x16(kptr + (size_t)s0 * D,             D);
    v16bf kb1 = load_b_32x16(kptr + (size_t)s0 * D + 32,        D);
    v16bf kb2 = load_b_32x16(kptr + (size_t)(s0 + 16) * D,      D);
    v16bf kb3 = load_b_32x16(kptr + (size_t)(s0 + 16) * D + 32, D);
    if (sb + 1 < nblk) {
      __builtin_prefetch(kptr + (size_t)(s0 + 32) * D, 0, 1);
      __builtin_prefetch(vptr + s0 + 32, 0, 1);
    }
#endif

    // S = Q K^T  (16 x 32 -> two 16x16 f32 tiles)
    v8f sL = {}, sR = {};
    sL = wmma_bf16(qa0, kb0, sL);
    sL = wmma_bf16(qa1, kb1, sL);
    sR = wmma_bf16(qa0, kb2, sR);
    sR = wmma_bf16(qa1, kb3, sR);

    // scale + causal mask + online softmax (row r lives in 16 lanes of a half)
    float p0[8], p1[8];
#pragma unroll
    for (int r = 0; r < 8; ++r) {
      const int t = Mbase + rowOff + r;
      float v0 = sL[r] * 0.125f;                   // D^-0.5
      float v1 = sR[r] * 0.125f;
      if (s0 + colBase > t)      v0 = -3.0e38f;
      if (s0 + 16 + colBase > t) v1 = -3.0e38f;
      float mx = fmaxf(v0, v1);
#pragma unroll
      for (int off = 1; off < 16; off <<= 1) mx = fmaxf(mx, __shfl_xor(mx, off, 32));
      const float mnew = fmaxf(m_i[r], mx);
      const float corr = __expf(m_i[r] - mnew);
      m_i[r] = mnew;
      p0[r] = __expf(v0 - mnew);
      p1[r] = __expf(v1 - mnew);
      l_i[r] = l_i[r] * corr + p0[r] + p1[r];      // per-lane partial row-sum
      o0[r] *= corr; o1[r] *= corr; o2[r] *= corr; o3[r] *= corr;
    }

    // P: C-layout regs -> LDS -> A-layout regs (16x32 bf16)
#pragma unroll
    for (int r = 0; r < 8; ++r) {
      myp[(rowOff + r) * 32 + colBase]      = (bf16_t)p0[r];
      myp[(rowOff + r) * 32 + 16 + colBase] = (bf16_t)p1[r];
    }
    asm volatile("s_wait_dscnt 0" ::: "memory");
    v16bf pa;
    {
      const int row  = lane & 15;
      const int half = lane >> 4;
      const bf16_t* pp = myp + row * 32 + half * 8;
      v8bf lo = *(const v8bf*)pp;
      v8bf hi = *(const v8bf*)(pp + 16);
#pragma unroll
      for (int i = 0; i < 8; ++i) { pa[i] = lo[i]; pa[i + 8] = hi[i]; }
    }

    // O += P * V
#if FLASH_ASYNC
    v16bf vb0 = load_b_32x16(vt + 0  * 32, 32);
    v16bf vb1 = load_b_32x16(vt + 16 * 32, 32);
    v16bf vb2 = load_b_32x16(vt + 32 * 32, 32);
    v16bf vb3 = load_b_32x16(vt + 48 * 32, 32);
#else
    v16bf vb0 = load_b_32x16(vptr + (size_t)0  * T + s0, T);
    v16bf vb1 = load_b_32x16(vptr + (size_t)16 * T + s0, T);
    v16bf vb2 = load_b_32x16(vptr + (size_t)32 * T + s0, T);
    v16bf vb3 = load_b_32x16(vptr + (size_t)48 * T + s0, T);
#endif
    o0 = wmma_bf16(pa, vb0, o0);
    o1 = wmma_bf16(pa, vb1, o1);
    o2 = wmma_bf16(pa, vb2, o2);
    o3 = wmma_bf16(pa, vb3, o3);
  }

  // finalize: reduce row sums across the 16-lane half, normalize, store
  bf16_t* dst = attnb + ((size_t)b * T + Mbase) * NCOL + (size_t)(g * HPG + h) * D;
#pragma unroll
  for (int r = 0; r < 8; ++r) {
    float tot = l_i[r];
#pragma unroll
    for (int off = 1; off < 16; off <<= 1) tot += __shfl_xor(tot, off, 32);
    const float inv = 1.0f / tot;
    const size_t ra = (size_t)(rowOff + r) * NCOL;
    dst[ra + 0  + colBase] = (bf16_t)(o0[r] * inv);
    dst[ra + 16 + colBase] = (bf16_t)(o1[r] * inv);
    dst[ra + 32 + colBase] = (bf16_t)(o2[r] * inv);
    dst[ra + 48 + colBase] = (bf16_t)(o3[r] * inv);
  }
}

// ---------------------------------------------------------------------------
// Kernel 3: output projection  out[B*T, C] = attnb[B*T, NCOL] @ Wp + bp  (fp32)
// Wpt stored transposed [C, NCOL]; register-pipelined K loop.
// ---------------------------------------------------------------------------
__global__ void __launch_bounds__(256)
out_proj_kernel(const bf16_t* __restrict__ attnb, const bf16_t* __restrict__ Wpt,
                const float* __restrict__ bp, float* __restrict__ out) {
  const int wave = threadIdx.x >> 5;
  const int lane = threadIdx.x & 31;
  const int tg   = blockIdx.x * 8 + wave;          // 8192 waves
  const int ntile = tg >> 9;                       // 0..15  (64-wide col tiles)
  const int mtile = tg & 511;                      // 0..511 (16-row tiles)

  const bf16_t* arow = attnb + (size_t)mtile * 16 * NCOL;
  const bf16_t* wrow = Wpt + (size_t)(ntile * 64) * NCOL;

  v8f acc0 = {}, acc1 = {}, acc2 = {}, acc3 = {};
  v16bf a  = load_a_16x32(arow, NCOL);
  v16bf b0 = load_b_32x16(wrow + (size_t)0  * NCOL, NCOL);
  v16bf b1 = load_b_32x16(wrow + (size_t)16 * NCOL, NCOL);
  v16bf b2 = load_b_32x16(wrow + (size_t)32 * NCOL, NCOL);
  v16bf b3 = load_b_32x16(wrow + (size_t)48 * NCOL, NCOL);
  for (int kc = 32; kc < NCOL; kc += 32) {
    v16bf an  = load_a_16x32(arow + kc, NCOL);
    v16bf bn0 = load_b_32x16(wrow + (size_t)0  * NCOL + kc, NCOL);
    v16bf bn1 = load_b_32x16(wrow + (size_t)16 * NCOL + kc, NCOL);
    v16bf bn2 = load_b_32x16(wrow + (size_t)32 * NCOL + kc, NCOL);
    v16bf bn3 = load_b_32x16(wrow + (size_t)48 * NCOL + kc, NCOL);
    acc0 = wmma_bf16(a, b0, acc0);
    acc1 = wmma_bf16(a, b1, acc1);
    acc2 = wmma_bf16(a, b2, acc2);
    acc3 = wmma_bf16(a, b3, acc3);
    a = an; b0 = bn0; b1 = bn1; b2 = bn2; b3 = bn3;
  }
  acc0 = wmma_bf16(a, b0, acc0);
  acc1 = wmma_bf16(a, b1, acc1);
  acc2 = wmma_bf16(a, b2, acc2);
  acc3 = wmma_bf16(a, b3, acc3);

  const int colBase = lane & 15;
  const int rowOff  = (lane >> 4) << 3;
  const int c0 = ntile * 64;
  const float bi0 = bp[c0 + 0  + colBase];
  const float bi1 = bp[c0 + 16 + colBase];
  const float bi2 = bp[c0 + 32 + colBase];
  const float bi3 = bp[c0 + 48 + colBase];
#pragma unroll
  for (int r = 0; r < 8; ++r) {
    const size_t ra = (size_t)(mtile * 16 + rowOff + r) * C;
    out[ra + c0 + 0  + colBase] = acc0[r] + bi0;
    out[ra + c0 + 16 + colBase] = acc1[r] + bi1;
    out[ra + c0 + 32 + colBase] = acc2[r] + bi2;
    out[ra + c0 + 48 + colBase] = acc3[r] + bi3;
  }
}

// ---------------------------------------------------------------------------
// Host launcher
// ---------------------------------------------------------------------------
extern "C" void kernel_launch(void* const* d_in, const int* in_sizes, int n_in,
                              void* d_out, int out_size, void* d_ws, size_t ws_size,
                              hipStream_t stream) {
  const float* x  = (const float*)d_in[0];
  const float* Wq = (const float*)d_in[1];
  const float* Wk = (const float*)d_in[2];
  const float* Wv = (const float*)d_in[3];
  const float* Wp = (const float*)d_in[4];
  const float* bp = (const float*)d_in[5];
  float* out = (float*)d_out;

  char* ws = (char*)d_ws;
  bf16_t* xb  = (bf16_t*)ws;  ws += (size_t)B * T * C * 2;
  bf16_t* Wqt = (bf16_t*)ws;  ws += (size_t)G * HPG * C * D * 2;
  bf16_t* Wkt = (bf16_t*)ws;  ws += (size_t)G * C * D * 2;
  bf16_t* Wvt = (bf16_t*)ws;  ws += (size_t)G * C * D * 2;
  bf16_t* Wpt = (bf16_t*)ws;  ws += (size_t)NCOL * C * 2;
  bf16_t* Qb  = (bf16_t*)ws;  ws += (size_t)B * G * HPG * T * D * 2;
  bf16_t* Kb  = (bf16_t*)ws;  ws += (size_t)B * G * T * D * 2;
  bf16_t* Vt  = (bf16_t*)ws;  ws += (size_t)B * G * T * D * 2;
  bf16_t* attnb = (bf16_t*)ws;

  const int nx = B * T * C;
  cvt_kernel<<<(nx + 255) / 256, 256, 0, stream>>>(x, xb, nx);

  const int nWq = G * HPG * C * D;
  cvt_tr_kernel<<<(nWq + 255) / 256, 256, 0, stream>>>(Wq, Wqt, G * HPG, C, D);
  const int nWk = G * C * D;
  cvt_tr_kernel<<<(nWk + 255) / 256, 256, 0, stream>>>(Wk, Wkt, G, C, D);
  cvt_tr_kernel<<<(nWk + 255) / 256, 256, 0, stream>>>(Wv, Wvt, G, C, D);
  const int nWp = NCOL * C;
  cvt_tr_kernel<<<(nWp + 255) / 256, 256, 0, stream>>>(Wp, Wpt, 1, NCOL, C);

  // 12288 tiles / 8 waves per block
  qkv_proj_kernel<<<1536, 256, 0, stream>>>(xb, Wqt, Wkt, Wvt, Qb, Kb, Vt);

  // 8192 query tiles / 8 waves per block (dynamic LDS: double-buffered K/V + P)
  flash_attn_kernel<<<1024, 256, FLASH_SMEM_BYTES, stream>>>(Qb, Kb, Vt, attnb);

  // 8192 output tiles / 8 waves per block
  out_proj_kernel<<<1024, 256, 0, stream>>>(attnb, Wpt, bp, out);
}
